// NequIP_7275674599679
// MI455X (gfx1250) — matrix operand order repaired and checked by reading, not compiled, not run on hardware
//
#include <hip/hip_runtime.h>
#include <hip/hip_bf16.h>

#define N_NODES 50000
#define N_EDGES 1600000
#define NLAYERS 5
#define CUTOFF_R 5.0f
#define PI_F 3.14159265358979323846f

typedef _Float16 half_t;
typedef __attribute__((ext_vector_type(16))) _Float16 v16h;
typedef __attribute__((ext_vector_type(8)))  float    v8f;

__device__ __forceinline__ float silu_f(float x) { return x / (1.0f + expf(-x)); }

__device__ __forceinline__ v8f wmma16(v16h a, v16h b, v8f c) {
  // v_wmma_f32_16x16x32_f16: D = A(16x32 f16) x B(32x16 f16) + C(16x16 f32)
  return __builtin_amdgcn_wmma_f32_16x16x32_f16(false, a, false, b, (short)0, c,
                                                false, false);
}

// A-fragment (16x32 f16) from LDS tile [16 x stride] row-major, K-chunk kc.
// ISA 7.12.2: lanes 0-15 M=lane, lanes 16-31 M=lane-16;
// half t -> K = (t<8 ? t : t+8) + (lane>=16 ? 8 : 0)
// Per-lane halves are two contiguous 8-half runs -> compiler emits ds_load_b128 x2.
__device__ __forceinline__ v16h a_frag(const half_t* tile, int stride, int kc, int lane) {
  const int m = lane & 15;
  const int kofs = kc * 32 + ((lane >> 4) << 3);
  v16h a;
#pragma unroll
  for (int t = 0; t < 16; ++t) {
    int K = kofs + (t < 8 ? t : t + 8);
    a[t] = tile[m * stride + K];
  }
  return a;
}

// B-fragment from PRE-SWIZZLED weight buffer: layout [kc][nt][lane][16 halves].
// One aligned 32-byte v16h read per fragment (2x global_load_b128, no packing VALU).
__device__ __forceinline__ v16h b_frag_packed(const half_t* P, int nn, int kc, int nt,
                                              int lane) {
  const v16h* p = (const v16h*)P;
  return p[(kc * nn + nt) * 32 + lane];
}

// ---------------------------------------------------------------- prep
// Packs a K-major f32 weight W[K][Ncols] into fragment order:
// dst[((kc*nn + nt)*32 + lane)*16 + t] = W[kc*32 + 8*(lane>=16) + (t<8 ? t : t+8)][nt*16 + lane%16]
__device__ __forceinline__ void pack_b(const float* __restrict__ W, int K, int Ncols,
                                       half_t* __restrict__ dst) {
  const int nn = Ncols >> 4;
  const int total = (K >> 5) * nn * 32 * 16;
  for (int i = threadIdx.x; i < total; i += blockDim.x) {
    int t = i & 15;
    int lane = (i >> 4) & 31;
    int nt = (i >> 9) % nn;
    int kc = (i >> 9) / nn;
    int k = kc * 32 + ((lane >> 4) << 3) + (t < 8 ? t : t + 8);
    int n = nt * 16 + (lane & 15);
    dst[i] = (half_t)W[k * Ncols + n];
  }
}

__global__ void prep_weights(const float* __restrict__ rad_w2, const float* __restrict__ rad_b2,
                             const float* __restrict__ self_w, const float* __restrict__ proj_w,
                             const float* __restrict__ mlp_w1, const float* __restrict__ mlp_w2,
                             half_t* __restrict__ w2s, float* __restrict__ b2s,
                             half_t* __restrict__ selfh, half_t* __restrict__ projh,
                             half_t* __restrict__ m1h, half_t* __restrict__ m2h) {
  const int l = blockIdx.x;
  const int tid = threadIdx.x;
  // W2s[k][n] = sum_l rad_w2[k][3n+l] (folds the lmax sum), packed fragment order.
  for (int i = tid; i < 2 * 4 * 32 * 16; i += blockDim.x) {
    int t = i & 15;
    int lane = (i >> 4) & 31;
    int nt = (i >> 9) & 3;
    int kc = i >> 11;
    int k = kc * 32 + ((lane >> 4) << 3) + (t < 8 ? t : t + 8);
    int n = nt * 16 + (lane & 15);
    const float* src = rad_w2 + l * 64 * 192 + k * 192 + n * 3;
    w2s[l * 4096 + i] = (half_t)(src[0] + src[1] + src[2]);
  }
  if (tid < 64) {
    const float* sb = rad_b2 + l * 192 + tid * 3;
    b2s[l * 64 + tid] = sb[0] + sb[1] + sb[2];
  }
  pack_b(self_w + l * 4096, 64, 64, selfh + l * 4096);
  pack_b(proj_w + l * 8192, 128, 64, projh + l * 8192);
  pack_b(mlp_w1 + l * 8192, 64, 128, m1h + l * 8192);
  pack_b(mlp_w2 + l * 8192, 128, 64, m2h + l * 8192);
}

// ---------------------------------------------------------------- RBF (layer-invariant)
__global__ void rbf_kernel(const float* __restrict__ pos, const int* __restrict__ row,
                           const int* __restrict__ col, const float* __restrict__ widths,
                           float4* __restrict__ rbf4) {
  int e = blockIdx.x * blockDim.x + threadIdx.x;
  if (e >= N_EDGES) return;
  int i = row[e], j = col[e];
  float dx = pos[j * 3 + 0] - pos[i * 3 + 0];
  float dy = pos[j * 3 + 1] - pos[i * 3 + 1];
  float dz = pos[j * 3 + 2] - pos[i * 3 + 2];
  float len = sqrtf(dx * dx + dy * dy + dz * dz);
  float cut = 0.5f * (cosf(len * (PI_F / CUTOFF_R)) + 1.0f);
  cut *= (len < CUTOFF_R) ? 1.0f : 0.0f;
  float r[8];
#pragma unroll
  for (int k = 0; k < 8; ++k) {
    float c = CUTOFF_R * (float)k / 7.0f;            // linspace(0,5,8)
    float w = fmaxf(widths[k], 0.1f);
    float d = (len - c) / w;
    r[k] = expf(-0.5f * d * d) * cut;
  }
  rbf4[e * 2 + 0] = make_float4(r[0], r[1], r[2], r[3]);
  rbf4[e * 2 + 1] = make_float4(r[4], r[5], r[6], r[7]);
}

// ---------------------------------------------------------------- initial embedding
__global__ void embed_kernel(const int* __restrict__ z, const float* __restrict__ embed,
                             float* __restrict__ feats) {
  int i = blockIdx.x * blockDim.x + threadIdx.x;   // grid sized to exactly N*64
  feats[i] = embed[z[i >> 6] * 64 + (i & 63)];
}

// ---------------------------------------------------------------- edge conv (the big one)
__global__ void __launch_bounds__(256) edge_conv(
    const float4* __restrict__ rbf4, const int* __restrict__ row, const int* __restrict__ col,
    const float* __restrict__ feats, float* __restrict__ agg,
    const float* __restrict__ W1, const float* __restrict__ B1,
    const half_t* __restrict__ W2s, const float* __restrict__ B2s) {
  __shared__ float sW1[8 * 64];
  __shared__ float sB1[64];
  __shared__ half_t sH[8][16 * 64];
  const int tid = threadIdx.x;
  const int lane = tid & 31, wave = tid >> 5;
  for (int i = tid; i < 512; i += 256) sW1[i] = W1[i];
  if (tid < 64) sB1[tid] = B1[tid];
  __syncthreads();

  const int e0 = (blockIdx.x * 8 + wave) * 16;
  {
    const int m = lane & 15;
    const int c0 = (lane >> 4) * 32;
    const int e = e0 + m;
    float4 rlo = rbf4[e * 2 + 0];
    float4 rhi = rbf4[e * 2 + 1];
    float r[8] = {rlo.x, rlo.y, rlo.z, rlo.w, rhi.x, rhi.y, rhi.z, rhi.w};
#pragma unroll
    for (int cc = 0; cc < 32; ++cc) {
      int c = c0 + cc;
      float acc = sB1[c];
#pragma unroll
      for (int k = 0; k < 8; ++k) acc += r[k] * sW1[k * 64 + c];
      sH[wave][m * 64 + c] = (half_t)silu_f(acc);
    }
  }
  __syncthreads();

  const int mb = (lane >> 4) * 8;
  const int nlo = lane & 15;
  int er[8], ec[8];
#pragma unroll
  for (int r = 0; r < 8; ++r) {
    er[r] = row[e0 + mb + r] * 64;   // pre-scaled row offsets (fit in int)
    ec[r] = col[e0 + mb + r] * 64;
  }

  // Batch all gathers up-front: 32 independent loads in flight at once.
  float fv[4][8];
#pragma unroll
  for (int nt = 0; nt < 4; ++nt) {
    const int n = nt * 16 + nlo;
#pragma unroll
    for (int r = 0; r < 8; ++r) fv[nt][r] = feats[ec[r] + n];
  }

  v16h a0 = a_frag(sH[wave], 64, 0, lane);
  v16h a1 = a_frag(sH[wave], 64, 1, lane);
  v8f cacc[4];
#pragma unroll
  for (int nt = 0; nt < 4; ++nt) {
    v8f c = {};
    c = wmma16(a0, b_frag_packed(W2s, 4, 0, nt, lane), c);
    c = wmma16(a1, b_frag_packed(W2s, 4, 1, nt, lane), c);
    cacc[nt] = c;
  }

#pragma unroll
  for (int nt = 0; nt < 4; ++nt) {
    const int n = nt * 16 + nlo;
    const float bn = B2s[n];
#pragma unroll
    for (int r = 0; r < 8; ++r) {
      // C layout: VGPR r holds row M = r + 8*(lane>=16), col N = lane%16
      float val = (cacc[nt][r] + bn) * fv[nt][r];
      atomicAdd(&agg[er[r] + n], val);
    }
  }
}

// ---------------------------------------------------------------- node update
__global__ void __launch_bounds__(128) node_update(
    float* __restrict__ feats, const float* __restrict__ agg,
    const half_t* __restrict__ Wself, const float* __restrict__ Bself,
    const half_t* __restrict__ Wproj, const float* __restrict__ Bproj,
    const half_t* __restrict__ Wm1, const float* __restrict__ Bm1,
    const half_t* __restrict__ Wm2, const float* __restrict__ Bm2,
    const float* __restrict__ lng, const float* __restrict__ lnb) {
  __shared__ half_t sF[4][1024];
  __shared__ half_t sG[4][1024];
  __shared__ half_t sS[4][1024];
  __shared__ half_t sT[4][2048];
  __shared__ float  sR[4][1024];
  const int tid = threadIdx.x, lane = tid & 31, wave = tid >> 5;
  const int n0 = (blockIdx.x * 4 + wave) * 16;
  const int lm = lane & 15;
  const int c0 = (lane >> 4) * 32;
  const int nid = (n0 + lm < N_NODES) ? (n0 + lm) : (N_NODES - 1);

#pragma unroll
  for (int cc = 0; cc < 32; ++cc) {
    int c = c0 + cc;
    float fv = feats[nid * 64 + c];
    float gv = agg[nid * 64 + c];
    sF[wave][lm * 64 + c] = (half_t)fv;
    sG[wave][lm * 64 + c] = (half_t)gv;
    sR[wave][lm * 64 + c] = fv;                    // residual stash (fp32)
  }
  __syncthreads();

  v16h aF0 = a_frag(sF[wave], 64, 0, lane), aF1 = a_frag(sF[wave], 64, 1, lane);
  v16h aG0 = a_frag(sG[wave], 64, 0, lane), aG1 = a_frag(sG[wave], 64, 1, lane);
  const int nlo = lane & 15, mhi = (lane >> 4) * 8;

  // self_out = feats @ self_w + self_b
#pragma unroll
  for (int nt = 0; nt < 4; ++nt) {
    v8f c = {};
    c = wmma16(aF0, b_frag_packed(Wself, 4, 0, nt, lane), c);
    c = wmma16(aF1, b_frag_packed(Wself, 4, 1, nt, lane), c);
    int n = nt * 16 + nlo;
    float bb = Bself[n];
#pragma unroll
    for (int r = 0; r < 8; ++r) sS[wave][(mhi + r) * 64 + n] = (half_t)(c[r] + bb);
  }
  __syncthreads();
  v16h aS0 = a_frag(sS[wave], 64, 0, lane), aS1 = a_frag(sS[wave], 64, 1, lane);
  __syncthreads();

  // conv = self_out @ proj[0:64] + agg @ proj[64:128] + b   (concat fused as 2 GEMMs)
#pragma unroll
  for (int nt = 0; nt < 4; ++nt) {
    v8f c = {};
    c = wmma16(aS0, b_frag_packed(Wproj, 4, 0, nt, lane), c);
    c = wmma16(aS1, b_frag_packed(Wproj, 4, 1, nt, lane), c);
    c = wmma16(aG0, b_frag_packed(Wproj, 4, 2, nt, lane), c);
    c = wmma16(aG1, b_frag_packed(Wproj, 4, 3, nt, lane), c);
    int n = nt * 16 + nlo;
    float bb = Bproj[n];
#pragma unroll
    for (int r = 0; r < 8; ++r) sS[wave][(mhi + r) * 64 + n] = (half_t)(c[r] + bb);
  }
  __syncthreads();
  v16h aC0 = a_frag(sS[wave], 64, 0, lane), aC1 = a_frag(sS[wave], 64, 1, lane);

  // t = silu(conv @ mlp_w1 + b1)  -> [16,128]
#pragma unroll
  for (int nt = 0; nt < 8; ++nt) {
    v8f c = {};
    c = wmma16(aC0, b_frag_packed(Wm1, 8, 0, nt, lane), c);
    c = wmma16(aC1, b_frag_packed(Wm1, 8, 1, nt, lane), c);
    int n = nt * 16 + nlo;
    float bb = Bm1[n];
#pragma unroll
    for (int r = 0; r < 8; ++r)
      sT[wave][(mhi + r) * 128 + n] = (half_t)silu_f(c[r] + bb);
  }
  __syncthreads();
  v16h aT0 = a_frag(sT[wave], 128, 0, lane), aT1 = a_frag(sT[wave], 128, 1, lane);
  v16h aT2 = a_frag(sT[wave], 128, 2, lane), aT3 = a_frag(sT[wave], 128, 3, lane);

  // upd = t @ mlp_w2 + b2 ; residual add in fp32
#pragma unroll
  for (int nt = 0; nt < 4; ++nt) {
    v8f c = {};
    c = wmma16(aT0, b_frag_packed(Wm2, 4, 0, nt, lane), c);
    c = wmma16(aT1, b_frag_packed(Wm2, 4, 1, nt, lane), c);
    c = wmma16(aT2, b_frag_packed(Wm2, 4, 2, nt, lane), c);
    c = wmma16(aT3, b_frag_packed(Wm2, 4, 3, nt, lane), c);
    int n = nt * 16 + nlo;
    float bb = Bm2[n];
#pragma unroll
    for (int r = 0; r < 8; ++r) sR[wave][(mhi + r) * 64 + n] += c[r] + bb;
  }
  __syncthreads();

  // LayerNorm: lanes 0..15 each own one node row
  if (lane < 16) {
    const int node = n0 + lane;
    const float* xr = sR[wave] + lane * 64;
    float mu = 0.f;
#pragma unroll
    for (int c = 0; c < 64; ++c) mu += xr[c];
    mu *= (1.0f / 64.0f);
    float var = 0.f;
#pragma unroll
    for (int c = 0; c < 64; ++c) { float d = xr[c] - mu; var += d * d; }
    var *= (1.0f / 64.0f);
    float inv = rsqrtf(var + 1e-5f);
    if (node < N_NODES) {
#pragma unroll
      for (int c = 0; c < 64; ++c)
        feats[node * 64 + c] = (xr[c] - mu) * inv * lng[c] + lnb[c];
    }
  }
}

// ---------------------------------------------------------------- readout (tiny)
__global__ void __launch_bounds__(256) readout(
    const float* __restrict__ feats, const int* __restrict__ z,
    const float* __restrict__ w1, const float* __restrict__ b1,
    const float* __restrict__ w2, const float* __restrict__ b2,
    const float* __restrict__ w3, const float* __restrict__ b3,
    const float* __restrict__ atom_e, float* __restrict__ out) {
  __shared__ float red[256];
  int n = blockIdx.x * 256 + threadIdx.x;
  float e = 0.f;
  if (n < N_NODES) {
    float f[64];
#pragma unroll
    for (int k = 0; k < 64; ++k) f[k] = feats[n * 64 + k];
    float h2[32];
#pragma unroll
    for (int j = 0; j < 32; ++j) h2[j] = b2[j];
    for (int j = 0; j < 64; ++j) {
      float a = b1[j];
#pragma unroll
      for (int k = 0; k < 64; ++k) a += f[k] * w1[k * 64 + j];
      a = silu_f(a);
#pragma unroll
      for (int jj = 0; jj < 32; ++jj) h2[jj] += a * w2[j * 32 + jj];
    }
    e = b3[0] + atom_e[z[n]];
#pragma unroll
    for (int jj = 0; jj < 32; ++jj) e += silu_f(h2[jj]) * w3[jj];
  }
  red[threadIdx.x] = e;
  __syncthreads();
  for (int s = 128; s > 0; s >>= 1) {
    if (threadIdx.x < s) red[threadIdx.x] += red[threadIdx.x + s];
    __syncthreads();
  }
  if (threadIdx.x == 0) atomicAdd(out, red[0]);
}

// ---------------------------------------------------------------- launch
extern "C" void kernel_launch(void* const* d_in, const int* in_sizes, int n_in,
                              void* d_out, int out_size, void* d_ws, size_t ws_size,
                              hipStream_t stream) {
  const int*   z      = (const int*)d_in[0];
  const float* pos    = (const float*)d_in[1];
  const int*   ei     = (const int*)d_in[2];
  const float* widths = (const float*)d_in[3];
  const float* embed  = (const float*)d_in[4];
  const float* rad_w1 = (const float*)d_in[5];
  const float* rad_b1 = (const float*)d_in[6];
  const float* rad_w2 = (const float*)d_in[7];
  const float* rad_b2 = (const float*)d_in[8];
  const float* self_w = (const float*)d_in[9];
  const float* self_b = (const float*)d_in[10];
  const float* proj_w = (const float*)d_in[11];
  const float* proj_b = (const float*)d_in[12];
  const float* mlp_w1 = (const float*)d_in[13];
  const float* mlp_b1 = (const float*)d_in[14];
  const float* mlp_w2 = (const float*)d_in[15];
  const float* mlp_b2 = (const float*)d_in[16];
  const float* ln_g   = (const float*)d_in[17];
  const float* ln_b   = (const float*)d_in[18];
  const float* ro_w1  = (const float*)d_in[19];
  const float* ro_b1  = (const float*)d_in[20];
  const float* ro_w2  = (const float*)d_in[21];
  const float* ro_b2  = (const float*)d_in[22];
  const float* ro_w3  = (const float*)d_in[23];
  const float* ro_b3  = (const float*)d_in[24];
  const float* atom_e = (const float*)d_in[25];
  (void)in_sizes; (void)n_in; (void)out_size; (void)ws_size;

  char* ws = (char*)d_ws;
  size_t off = 0;
  auto carve = [&](size_t bytes) -> void* {
    void* p = ws + off;
    off = (off + bytes + 255) & ~(size_t)255;
    return p;
  };
  float*  rbf   = (float*)carve((size_t)N_EDGES * 8 * 4);
  float*  feats = (float*)carve((size_t)N_NODES * 64 * 4);
  float*  agg   = (float*)carve((size_t)N_NODES * 64 * 4);
  half_t* w2s   = (half_t*)carve((size_t)NLAYERS * 4096 * 2);
  float*  b2s   = (float*)carve((size_t)NLAYERS * 64 * 4);
  half_t* selfh = (half_t*)carve((size_t)NLAYERS * 4096 * 2);
  half_t* projh = (half_t*)carve((size_t)NLAYERS * 8192 * 2);
  half_t* m1h   = (half_t*)carve((size_t)NLAYERS * 8192 * 2);
  half_t* m2h   = (half_t*)carve((size_t)NLAYERS * 8192 * 2);

  const int* erow = ei;
  const int* ecol = ei + N_EDGES;

  prep_weights<<<NLAYERS, 256, 0, stream>>>(rad_w2, rad_b2, self_w, proj_w, mlp_w1, mlp_w2,
                                            w2s, b2s, selfh, projh, m1h, m2h);
  rbf_kernel<<<N_EDGES / 256, 256, 0, stream>>>(pos, erow, ecol, widths, (float4*)rbf);
  embed_kernel<<<(N_NODES * 64) / 256, 256, 0, stream>>>(z, embed, feats);

  for (int l = 0; l < NLAYERS; ++l) {
    hipMemsetAsync(agg, 0, (size_t)N_NODES * 64 * 4, stream);
    edge_conv<<<N_EDGES / 128, 256, 0, stream>>>(
        (const float4*)rbf, erow, ecol, feats, agg,
        rad_w1 + l * 8 * 64, rad_b1 + l * 64,
        w2s + l * 4096, b2s + l * 64);
    node_update<<<(N_NODES + 63) / 64, 128, 0, stream>>>(
        feats, agg,
        selfh + l * 4096, self_b + l * 64,
        projh + l * 8192, proj_b + l * 64,
        m1h + l * 8192, mlp_b1 + l * 128,
        m2h + l * 8192, mlp_b2 + l * 64,
        ln_g + l * 64, ln_b + l * 64);
  }
  hipMemsetAsync(d_out, 0, sizeof(float), stream);
  readout<<<(N_NODES + 255) / 256, 256, 0, stream>>>(feats, z, ro_w1, ro_b1, ro_w2, ro_b2,
                                                     ro_w3, ro_b3, atom_e, (float*)d_out);
}
